// QuantWindowAttention_27049704030390
// MI455X (gfx1250) — matrix operand order repaired and checked by reading, not compiled, not run on hardware
//
#include <hip/hip_runtime.h>
#include <stdint.h>

typedef float    v4f  __attribute__((ext_vector_type(4)));
typedef float    v8f  __attribute__((ext_vector_type(8)));
typedef _Float16 v4h  __attribute__((ext_vector_type(4)));
typedef _Float16 v8h  __attribute__((ext_vector_type(8)));
typedef _Float16 v16h __attribute__((ext_vector_type(16)));

#define N_TOK 49
#define C_DIM 384
#define H_NUM 12
#define D_HEAD 32
#define NWIN 64
#define B_WIN 2048

// LDS row strides (in halves / floats). ldA16 rows need 16B align (stride mult 8 halves),
// ldB16 rows need 16B align too (stride mult 8 halves).
#define SXH 392  // x / proj-A hi,lo planes [64][SXH]  (A operand)
#define SQH 40   // q hi,lo planes          [64][SQH]  (A operand)
#define SKH 48   // k hi,lo planes          [64][SKH]  (B operand, 96B rows)
#define SVT 80   // v^T hi,lo planes        [32][SVT]  (B operand, 160B rows)
#define SP  72   // P hi,lo planes          [64][SP]   (A operand)
#define SS  68   // scores f32              [64][SS]
#define SO  388  // out accum f32           [64][SO]  (also async staging buffer)

// ---- WMMA / operand helpers -------------------------------------------------
__device__ __forceinline__ v8f wmma16(v16h a, v16h b, v8f c) {
  return __builtin_amdgcn_wmma_f32_16x16x32_f16(false, a, false, b, (short)0, c, false, false);
}
// A operand (16x32 f16): lane reads K[8*hi..8*hi+8) and K[16+8*hi..16+8*hi+8)
__device__ __forceinline__ v16h ldA16(const _Float16* rowbase, int kc, int hi) {
  v8h a = *(const v8h*)(rowbase + kc + 8 * hi);
  v8h b = *(const v8h*)(rowbase + kc + 16 + 8 * hi);
  return __builtin_shufflevector(a, b, 0,1,2,3,4,5,6,7,8,9,10,11,12,13,14,15);
}
// B operand (32x16 f16): lane reads K[8*hi..8*hi+8) and K[16+8*hi..16+8*hi+8)
__device__ __forceinline__ v16h ldB16(const _Float16* rowbase, int kc, int hi) {
  v8h a = *(const v8h*)(rowbase + kc + 8 * hi);
  v8h b = *(const v8h*)(rowbase + kc + 16 + 8 * hi);
  return __builtin_shufflevector(a, b, 0,1,2,3,4,5,6,7,8,9,10,11,12,13,14,15);
}
__device__ __forceinline__ void split2(float v, _Float16& hs, _Float16& ls) {
  _Float16 hv = (_Float16)v;
  hs = hv;
  ls = (_Float16)(v - (float)hv);
}

// ---- prep: combined bias  cb[w][h][m][n] = rpb[rel[m*49+n]][h] + mask[w][m][n]
__global__ void bias_combine_kernel(const float* __restrict__ mask,
                                    const float* __restrict__ rpb,
                                    const int*   __restrict__ rel,
                                    float* __restrict__ cb, int total) {
  int i = blockIdx.x * blockDim.x + threadIdx.x;
  if (i >= total) return;
  int n = i % N_TOK;
  int m = (i / N_TOK) % N_TOK;
  int h = (i / (N_TOK * N_TOK)) % H_NUM;
  int w = i / (N_TOK * N_TOK * H_NUM);
  cb[i] = rpb[rel[m * N_TOK + n] * H_NUM + h] + mask[(w * N_TOK + m) * N_TOK + n];
}

// ---- prep: split fp32 weights into hi/lo f16 planes
__global__ void split_w_kernel(const float* __restrict__ w,
                               _Float16* __restrict__ wh,
                               _Float16* __restrict__ wl, int total) {
  int i = blockIdx.x * blockDim.x + threadIdx.x;
  if (i >= total) return;
  split2(w[i], wh[i], wl[i]);
}

__launch_bounds__(256, 1)
__global__ void win_attn_kernel(const float* __restrict__ x,
                                const _Float16* __restrict__ qwh,
                                const _Float16* __restrict__ qwl,
                                const _Float16* __restrict__ pwh,
                                const _Float16* __restrict__ pwl,
                                const float* __restrict__ qkv_b,
                                const float* __restrict__ proj_b,
                                const float* __restrict__ cb,
                                float* __restrict__ out,
                                float* __restrict__ attn_out) {
  __shared__ __align__(32) _Float16 lds_xh [64 * SXH];
  __shared__ __align__(32) _Float16 lds_xl [64 * SXH];
  __shared__ __align__(32) _Float16 lds_qh [64 * SQH];
  __shared__ __align__(32) _Float16 lds_ql [64 * SQH];
  __shared__ __align__(32) _Float16 lds_kh [64 * SKH];
  __shared__ __align__(32) _Float16 lds_kl [64 * SKH];
  __shared__ __align__(32) _Float16 lds_vth[32 * SVT];
  __shared__ __align__(32) _Float16 lds_vtl[32 * SVT];
  __shared__ __align__(32) _Float16 lds_Ph [64 * SP];
  __shared__ __align__(32) _Float16 lds_Pl [64 * SP];
  __shared__ __align__(16) float    lds_S  [64 * SS];
  __shared__ __align__(16) float    lds_o  [64 * SO];  // P.V accum; staging buffer early

  const int b    = blockIdx.x;
  const int tid  = threadIdx.x;
  const int wave = tid >> 5;
  const int lane = tid & 31;
  const int ln   = lane & 15;
  const int hi   = lane >> 4;
  const float scale = 0.17677669529663687f;  // 32^-0.5

  // ---- stage raw x window into lds_o via CDNA5 async LDS loads (ASYNCcnt path)
  {
    const float* xp = x + (size_t)b * N_TOK * C_DIM;
    for (int i = tid; i < N_TOK * (C_DIM / 4); i += 256) {
      uint32_t ldsa = (uint32_t)(uintptr_t)(&lds_o[i * 4]);  // LDS byte offset (addr[31:0])
      uint32_t goff = (uint32_t)(i * 16);
      asm volatile("global_load_async_to_lds_b128 %0, %1, %2 offset:0"
                   :: "v"(ldsa), "v"(goff), "s"(xp) : "memory");
    }
    asm volatile("s_wait_asynccnt 0" ::: "memory");
    __syncthreads();
    // convert to f16 hi/lo planes; zero the 15 padding rows
    for (int i = tid; i < N_TOK * (C_DIM / 4); i += 256) {
      int r = i / (C_DIM / 4), c4 = (i % (C_DIM / 4)) * 4;
      v4f v = *(const v4f*)(&lds_o[i * 4]);
      v4h hp, lp;
#pragma unroll
      for (int j = 0; j < 4; ++j) { _Float16 hh, ll; split2(v[j], hh, ll); hp[j] = hh; lp[j] = ll; }
      *(v4h*)(&lds_xh[r * SXH + c4]) = hp;
      *(v4h*)(&lds_xl[r * SXH + c4]) = lp;
    }
    v4h z = {};
    for (int i = tid; i < 15 * (C_DIM / 4); i += 256) {
      int r = 49 + i / (C_DIM / 4), c4 = (i % (C_DIM / 4)) * 4;
      *(v4h*)(&lds_xh[r * SXH + c4]) = z;
      *(v4h*)(&lds_xl[r * SXH + c4]) = z;
    }
  }
  __syncthreads();

  const float* cbw = cb + (size_t)(b & (NWIN - 1)) * H_NUM * N_TOK * N_TOK;

  for (int h = 0; h < H_NUM; ++h) {
    // ======== phase 1: q/k/v for this head — split-f16 WMMA (24 tiles, 3/wave, joint)
    {
      int mats[3], colsg[3], mt_[3], nt_[3];
      const _Float16 *axh[3], *axl[3], *bwh[3], *bwl[3];
#pragma unroll
      for (int i = 0; i < 3; ++i) {
        int t = wave * 3 + i;
        int mat = t >> 3, r = t & 7, nt = r >> 2, mt = r & 3;
        int ncol = mat * C_DIM + h * D_HEAD + nt * 16;
        mats[i] = mat; mt_[i] = mt; nt_[i] = nt; colsg[i] = ncol;
        axh[i] = lds_xh + (mt * 16 + ln) * SXH;
        axl[i] = lds_xl + (mt * 16 + ln) * SXH;
        bwh[i] = qwh + (size_t)(ncol + ln) * C_DIM;
        bwl[i] = qwl + (size_t)(ncol + ln) * C_DIM;
      }
      v8f acc[3] = {};
#pragma unroll 2
      for (int kc = 0; kc < C_DIM; kc += 32) {
#pragma unroll
        for (int i = 0; i < 3; ++i) {
          v16h Ah = ldA16(axh[i], kc, hi);
          v16h Al = ldA16(axl[i], kc, hi);
          v16h Bh = ldB16(bwh[i], kc, hi);
          v16h Bl = ldB16(bwl[i], kc, hi);
          acc[i] = wmma16(Ah, Bh, acc[i]);
          acc[i] = wmma16(Al, Bh, acc[i]);
          acc[i] = wmma16(Ah, Bl, acc[i]);
        }
      }
#pragma unroll
      for (int i = 0; i < 3; ++i) {
        float bias = qkv_b[colsg[i] + ln];
        int col = nt_[i] * 16 + ln;
        if (mats[i] == 0) {        // q: pre-scaled, hi/lo planes (A operand later)
#pragma unroll
          for (int j = 0; j < 8; ++j) {
            int m = mt_[i] * 16 + j + 8 * hi;
            split2((acc[i][j] + bias) * scale, lds_qh[m * SQH + col], lds_ql[m * SQH + col]);
          }
        } else if (mats[i] == 1) { // k: hi/lo planes (B operand later)
#pragma unroll
          for (int j = 0; j < 8; ++j) {
            int m = mt_[i] * 16 + j + 8 * hi;
            split2(acc[i][j] + bias, lds_kh[m * SKH + col], lds_kl[m * SKH + col]);
          }
        } else {                   // v: transposed [dv][token] hi/lo planes (B operand)
#pragma unroll
          for (int j = 0; j < 8; ++j) {
            int m = mt_[i] * 16 + j + 8 * hi;
            split2(acc[i][j] + bias, lds_vth[col * SVT + m], lds_vtl[col * SVT + m]);
          }
        }
      }
    }
    __syncthreads();

    // ======== phase 2: S = (q*scale)·k^T + bias+mask — split-f16 WMMA (2 tiles/wave)
    {
      const float* cbh = cbw + (size_t)h * N_TOK * N_TOK;
#pragma unroll
      for (int i = 0; i < 2; ++i) {
        int t = wave * 2 + i;
        int mtile = t >> 2, ntile = t & 3;
        v16h Ah = ldA16(lds_qh + (mtile * 16 + ln) * SQH, 0, hi);
        v16h Al = ldA16(lds_ql + (mtile * 16 + ln) * SQH, 0, hi);
        v16h Bh = ldB16(lds_kh + (ntile * 16 + ln) * SKH, 0, hi);
        v16h Bl = ldB16(lds_kl + (ntile * 16 + ln) * SKH, 0, hi);
        v8f acc = {};
        acc = wmma16(Ah, Bh, acc);
        acc = wmma16(Al, Bh, acc);
        acc = wmma16(Ah, Bl, acc);
        int n = ntile * 16 + ln;
#pragma unroll
        for (int j = 0; j < 8; ++j) {
          int m = mtile * 16 + j + 8 * hi;
          float s = acc[j];
          if (m < N_TOK && n < N_TOK) s += cbh[m * N_TOK + n];
          lds_S[m * SS + n] = s;
        }
      }
    }
    __syncthreads();

    // ======== phase 3: row softmax -> P hi/lo planes (pad rows/cols zeroed); emit attn
    if (tid < N_TOK) {
      float* row = &lds_S[tid * SS];
      _Float16* ph = &lds_Ph[tid * SP];
      _Float16* pl = &lds_Pl[tid * SP];
      float mx = -1e30f;
      for (int n = 0; n < N_TOK; ++n) mx = fmaxf(mx, row[n]);
      float s = 0.f;
      for (int n = 0; n < N_TOK; ++n) { float e = __expf(row[n] - mx); row[n] = e; s += e; }
      float inv = 1.f / s;
      float* ao = attn_out + (((size_t)b * H_NUM + h) * N_TOK + tid) * N_TOK;
      for (int n = 0; n < N_TOK; ++n) {
        float p = row[n] * inv;
        ao[n] = p;
        split2(p, ph[n], pl[n]);
      }
      for (int n = N_TOK; n < 64; ++n) { ph[n] = (_Float16)0.f; pl[n] = (_Float16)0.f; }
    } else if (tid < 64) {  // zero pad rows of P so K=64 padding is exact
      _Float16* ph = &lds_Ph[tid * SP];
      _Float16* pl = &lds_Pl[tid * SP];
      for (int n = 0; n < 64; ++n) { ph[n] = (_Float16)0.f; pl[n] = (_Float16)0.f; }
    }
    __syncthreads();

    // ======== phase 4: out_h = P·V — split-f16 WMMA (1 tile/wave)
    {
      int mtile = wave >> 1, ntile = wave & 1;
      const _Float16* aph = lds_Ph + (mtile * 16 + ln) * SP;
      const _Float16* apl = lds_Pl + (mtile * 16 + ln) * SP;
      const _Float16* bvh = lds_vth + (ntile * 16 + ln) * SVT;
      const _Float16* bvl = lds_vtl + (ntile * 16 + ln) * SVT;
      v8f acc = {};
#pragma unroll
      for (int kc = 0; kc < 64; kc += 32) {
        v16h Ah = ldA16(aph, kc, hi);
        v16h Al = ldA16(apl, kc, hi);
        v16h Bh = ldB16(bvh, kc, hi);
        v16h Bl = ldB16(bvl, kc, hi);
        acc = wmma16(Ah, Bh, acc);
        acc = wmma16(Al, Bh, acc);
        acc = wmma16(Ah, Bl, acc);
      }
      int col = h * D_HEAD + ntile * 16 + ln;
#pragma unroll
      for (int j = 0; j < 8; ++j) {
        int m = mtile * 16 + j + 8 * hi;
        lds_o[m * SO + col] = acc[j];
      }
    }
    __syncthreads();
  }

  // ======== phase 5a: split f32 attention output into f16 hi/lo planes (reuse x planes)
  for (int i = tid; i < 64 * (C_DIM / 4); i += 256) {
    int r = i / (C_DIM / 4), c4 = (i % (C_DIM / 4)) * 4;
    v4f v = *(const v4f*)(&lds_o[r * SO + c4]);
    v4h hp, lp;
#pragma unroll
    for (int j = 0; j < 4; ++j) { _Float16 hh, ll; split2(v[j], hh, ll); hp[j] = hh; lp[j] = ll; }
    *(v4h*)(&lds_xh[r * SXH + c4]) = hp;
    *(v4h*)(&lds_xl[r * SXH + c4]) = lp;
  }
  __syncthreads();

  // ======== phase 5b: proj GEMM — split-f16 WMMA, 4 M-tiles share each B load
  for (int i = 0; i < 3; ++i) {
    int nt = wave * 3 + i;
    int ncol = nt * 16;
    const _Float16* bh = pwh + (size_t)(ncol + ln) * C_DIM;
    const _Float16* bl = pwl + (size_t)(ncol + ln) * C_DIM;
    const _Float16* ah0 = lds_xh + ( 0 + ln) * SXH;
    const _Float16* ah1 = lds_xh + (16 + ln) * SXH;
    const _Float16* ah2 = lds_xh + (32 + ln) * SXH;
    const _Float16* ah3 = lds_xh + (48 + ln) * SXH;
    const _Float16* al0 = lds_xl + ( 0 + ln) * SXH;
    const _Float16* al1 = lds_xl + (16 + ln) * SXH;
    const _Float16* al2 = lds_xl + (32 + ln) * SXH;
    const _Float16* al3 = lds_xl + (48 + ln) * SXH;
    v8f c0 = {}, c1 = {}, c2 = {}, c3 = {};
#pragma unroll 2
    for (int kc = 0; kc < C_DIM; kc += 32) {
      v16h Bh = ldB16(bh, kc, hi);
      v16h Bl = ldB16(bl, kc, hi);
      v16h A;
      A = ldA16(ah0, kc, hi); c0 = wmma16(A, Bh, c0); c0 = wmma16(A, Bl, c0);
      A = ldA16(al0, kc, hi); c0 = wmma16(A, Bh, c0);
      A = ldA16(ah1, kc, hi); c1 = wmma16(A, Bh, c1); c1 = wmma16(A, Bl, c1);
      A = ldA16(al1, kc, hi); c1 = wmma16(A, Bh, c1);
      A = ldA16(ah2, kc, hi); c2 = wmma16(A, Bh, c2); c2 = wmma16(A, Bl, c2);
      A = ldA16(al2, kc, hi); c2 = wmma16(A, Bh, c2);
      A = ldA16(ah3, kc, hi); c3 = wmma16(A, Bh, c3); c3 = wmma16(A, Bl, c3);
      A = ldA16(al3, kc, hi); c3 = wmma16(A, Bh, c3);
    }
    float pb = proj_b[ncol + ln];
    float* op = out + (size_t)b * N_TOK * C_DIM + ncol + ln;
#pragma unroll
    for (int j = 0; j < 8; ++j) {
      int m0 = j + 8 * hi;
      if (m0 < N_TOK)      op[(size_t)m0 * C_DIM]        = c0[j] + pb;
      if (m0 + 16 < N_TOK) op[(size_t)(m0 + 16) * C_DIM] = c1[j] + pb;
      if (m0 + 32 < N_TOK) op[(size_t)(m0 + 32) * C_DIM] = c2[j] + pb;
      if (m0 + 48 < N_TOK) op[(size_t)(m0 + 48) * C_DIM] = c3[j] + pb;
    }
  }
}

extern "C" void kernel_launch(void* const* d_in, const int* in_sizes, int n_in,
                              void* d_out, int out_size, void* d_ws, size_t ws_size,
                              hipStream_t stream) {
  const float* x      = (const float*)d_in[0];
  const float* mask   = (const float*)d_in[1];
  const float* qkv_w  = (const float*)d_in[2];
  const float* qkv_b  = (const float*)d_in[3];
  const float* proj_w = (const float*)d_in[4];
  const float* proj_b = (const float*)d_in[5];
  const float* rpb    = (const float*)d_in[6];
  const int*   rel    = (const int*)d_in[7];

  float* out  = (float*)d_out;                        // [2048,49,384]
  float* attn = out + (size_t)B_WIN * N_TOK * C_DIM;  // [2048,12,49,49]

  // workspace layout (all offsets 32B aligned)
  const int cb_elems = NWIN * H_NUM * N_TOK * N_TOK;      // 1,843,968 f32
  const int qw_elems = 3 * C_DIM * C_DIM;                 // 442,368
  const int pw_elems = C_DIM * C_DIM;                     // 147,456
  float*    cb  = (float*)d_ws;
  _Float16* qwh = (_Float16*)(cb + cb_elems);
  _Float16* qwl = qwh + qw_elems;
  _Float16* pwh = qwl + qw_elems;
  _Float16* pwl = pwh + pw_elems;

  bias_combine_kernel<<<(cb_elems + 255) / 256, 256, 0, stream>>>(mask, rpb, rel, cb, cb_elems);
  split_w_kernel<<<(qw_elems + 255) / 256, 256, 0, stream>>>(qkv_w, qwh, qwl, qw_elems);
  split_w_kernel<<<(pw_elems + 255) / 256, 256, 0, stream>>>(proj_w, pwh, pwl, pw_elems);
  win_attn_kernel<<<B_WIN, 256, 0, stream>>>(x, qwh, qwl, pwh, pwl,
                                             qkv_b, proj_b, cb, out, attn);
}
